// MultiheadAttention_39427799777793
// MI455X (gfx1250) — matrix operand order repaired and checked by reading, not compiled, not run on hardware
//
#include <hip/hip_runtime.h>
#include <cstdint>

typedef __attribute__((ext_vector_type(16))) __bf16 v16bf;
typedef __attribute__((ext_vector_type(8)))  __bf16 v8bf;
typedef __attribute__((ext_vector_type(8)))  float  v8f;

#define L_SEQ 1024
#define FDIM  1024
#define DHEAD 64

__device__ __forceinline__ v8f wmma_bf16(v16bf a, v16bf b, v8f c) {
  // D = A(16x32 bf16) x B(32x16 bf16) + C(16x16 f32)
  return __builtin_amdgcn_wmma_f32_16x16x32_bf16(false, a, false, b, (short)0, c,
                                                 false, false);
}

// gfx1250 async copy: 16B global -> LDS per lane, tracked by ASYNCcnt
__device__ __forceinline__ void async_ld_b128(void* lds_ptr, const void* gptr) {
  uint32_t lds_off = (uint32_t)(uintptr_t)lds_ptr;  // LDS aperture low 32 bits
  asm volatile("global_load_async_to_lds_b128 %0, %1, off"
               :: "v"(lds_off), "v"(gptr) : "memory");
}
__device__ __forceinline__ void wait_async0() {
  asm volatile("s_wait_asynccnt 0x0" ::: "memory");
}

// Build a 16-elem fragment from two contiguous 16B halves (ds/global b128 pairs)
__device__ __forceinline__ v16bf ld_frag16(const __bf16* p) {
  v8bf lo = *(const v8bf*)(p);
  v8bf hi = *(const v8bf*)(p + 8);
  return __builtin_shufflevector(lo, hi, 0, 1, 2, 3, 4, 5, 6, 7, 8, 9, 10, 11, 12,
                                 13, 14, 15);
}
// A-fragment (16-bit A 16x32): element e -> K = e + (e&8) + 8*half
__device__ __forceinline__ v16bf ld_afrag(const __bf16* row, int hf) {
  v8bf lo = *(const v8bf*)(row + hf * 8);
  v8bf hi = *(const v8bf*)(row + 16 + hf * 8);
  return __builtin_shufflevector(lo, hi, 0, 1, 2, 3, 4, 5, 6, 7, 8, 9, 10, 11, 12,
                                 13, 14, 15);
}
// B-fragment (16-bit B 32x16): element e -> K = e + (e&8) + 8*half at fixed column.
__device__ __forceinline__ v16bf ld_bfrag(const __bf16* colrow, int hf) {
  return ld_afrag(colrow, hf);
}

__device__ __forceinline__ uint32_t pack2_bf16(float a, float b) {
  union { __bf16 h[2]; uint32_t u; } x;
  x.h[0] = (__bf16)a;
  x.h[1] = (__bf16)b;
  return x.u;
}

__device__ __forceinline__ float hmax16(float x) {
  x = fmaxf(x, __shfl_xor(x, 1, 32));
  x = fmaxf(x, __shfl_xor(x, 2, 32));
  x = fmaxf(x, __shfl_xor(x, 4, 32));
  x = fmaxf(x, __shfl_xor(x, 8, 32));
  return x;
}
__device__ __forceinline__ float hsum16(float x) {
  x += __shfl_xor(x, 1, 32);
  x += __shfl_xor(x, 2, 32);
  x += __shfl_xor(x, 4, 32);
  x += __shfl_xor(x, 8, 32);
  return x;
}

// ---------------------------------------------------------------------------
// fp32 -> bf16 bulk convert (vectorized, one-time)
// ---------------------------------------------------------------------------
__global__ __launch_bounds__(256) void cvt_bf16_kernel(const float* __restrict__ in,
                                                       __bf16* __restrict__ out,
                                                       int n) {
  const int i = (blockIdx.x * 256 + threadIdx.x) * 4;
  if (i >= n) return;
  const float4 v = *(const float4*)(in + i);
  uint2 o;
  o.x = pack2_bf16(v.x, v.y);
  o.y = pack2_bf16(v.z, v.w);
  *(uint2*)(out + i) = o;
}

// ---------------------------------------------------------------------------
// W[K,N] fp32 -> Wt[N,K] bf16 (LDS tile transpose, one-time per weight)
// ---------------------------------------------------------------------------
__global__ __launch_bounds__(256) void transpose_bf16_kernel(
    const float* __restrict__ W, __bf16* __restrict__ Wt) {
  __shared__ float T[32][33];
  const int tx = threadIdx.x & 31, ty = threadIdx.x >> 5;  // 32x8
  const int n0 = blockIdx.x * 32, k0 = blockIdx.y * 32;
#pragma unroll
  for (int j = 0; j < 4; ++j)
    T[ty + 8 * j][tx] = W[(size_t)(k0 + ty + 8 * j) * FDIM + n0 + tx];
  __syncthreads();
#pragma unroll
  for (int j = 0; j < 4; ++j)
    Wt[(size_t)(n0 + ty + 8 * j) * FDIM + k0 + tx] = (__bf16)T[tx][ty + 8 * j];
}

// ---------------------------------------------------------------------------
// WMMA GEMM: C[M,N] = A[M,K](bf16) @ Wt[N,K](bf16, pre-transposed) + bias.
// 128x64 block tile, 4 waves, wave = 32x64 (8 WMMAs / K-step).
// Async double-buffered tile staging via global_load_async_to_lds_b128.
// MODE 0: bf16 [M,N]   MODE 1: bf16 V-transposed [b,h,d,L]   MODE 2: f32 [M,N]
// ---------------------------------------------------------------------------
template <int MODE>
__global__ __launch_bounds__(128) void gemm_bias_kernel(
    const __bf16* __restrict__ A, const __bf16* __restrict__ Bt_g,
    const float* __restrict__ bias, void* __restrict__ Cv, int M, int N, int K) {
  __shared__ __align__(16) __bf16 As[2][128][40];  // rows x 32K (+pad)
  __shared__ __align__(16) __bf16 Bs[2][64][40];   // cols x 32K (+pad)

  const int tid  = threadIdx.x;
  const int lane = tid & 31;
  const int w    = tid >> 5;
  const int r    = lane & 15;
  const int hf   = lane >> 4;
  const int nb   = N >> 6;
  const int n0   = (blockIdx.x % nb) * 64;
  const int m0   = (blockIdx.x / nb) * 128;

  v8f acc0[4] = {};
  v8f acc1[4] = {};

  auto issue_tiles = [&](int buf, int k0) {
    // A tile: 128 rows x 4 16B-chunks = 512 chunks, 4 per thread
#pragma unroll
    for (int it = 0; it < 4; ++it) {
      const int idx = it * 128 + tid;
      const int rr = idx >> 2, cc = idx & 3;
      async_ld_b128(&As[buf][rr][cc * 8], A + (size_t)(m0 + rr) * K + k0 + cc * 8);
    }
    // B tile: 64 rows x 4 chunks = 256 chunks, 2 per thread
#pragma unroll
    for (int it = 0; it < 2; ++it) {
      const int idx = it * 128 + tid;
      const int rr = idx >> 2, cc = idx & 3;
      async_ld_b128(&Bs[buf][rr][cc * 8],
                    Bt_g + (size_t)(n0 + rr) * K + k0 + cc * 8);
    }
  };

  int buf = 0;
  issue_tiles(0, 0);
#pragma unroll 1
  for (int k0 = 0; k0 < K; k0 += 32) {
    wait_async0();
    __syncthreads();
    if (k0 + 32 < K) issue_tiles(buf ^ 1, k0 + 32);

    const v16bf a0 = ld_afrag(&As[buf][w * 32 + r][0], hf);
    const v16bf a1 = ld_afrag(&As[buf][w * 32 + 16 + r][0], hf);
#pragma unroll
    for (int j = 0; j < 4; ++j) {
      const v16bf bf = ld_bfrag(&Bs[buf][j * 16 + r][0], hf);
      acc0[j] = wmma_bf16(a0, bf, acc0[j]);
      acc1[j] = wmma_bf16(a1, bf, acc1[j]);
    }
    buf ^= 1;
  }

#pragma unroll
  for (int j = 0; j < 4; ++j)
#pragma unroll
    for (int i = 0; i < 8; ++i) {
      const int col = n0 + j * 16 + r;
      const int row0 = m0 + w * 32 + i + 8 * hf;
      const int row1 = row0 + 16;
      const float v0 = acc0[j][i] + bias[col];
      const float v1 = acc1[j][i] + bias[col];
      if (MODE == 0) {
        ((__bf16*)Cv)[(size_t)row0 * N + col] = (__bf16)v0;
        ((__bf16*)Cv)[(size_t)row1 * N + col] = (__bf16)v1;
      } else if (MODE == 1) {  // V stored [b,h,d,L] for transposed attn tiles
        const int hh = col >> 6, dd = col & 63;
        const int b0 = row0 >> 10, l0g = row0 & 1023;
        const int b1 = row1 >> 10, l1g = row1 & 1023;
        ((__bf16*)Cv)[(((size_t)b0 * 16 + hh) * 64 + dd) * L_SEQ + l0g] = (__bf16)v0;
        ((__bf16*)Cv)[(((size_t)b1 * 16 + hh) * 64 + dd) * L_SEQ + l1g] = (__bf16)v1;
      } else {
        ((float*)Cv)[(size_t)row0 * N + col] = v0;
        ((float*)Cv)[(size_t)row1 * N + col] = v1;
      }
    }
}

// ---------------------------------------------------------------------------
// Ew = rel_embed[2049,64] @ rel_w[64,64]  -> bf16 (tiny)
// ---------------------------------------------------------------------------
__global__ void ew_kernel(const float* __restrict__ rel_embed,
                          const float* __restrict__ rel_w, __bf16* __restrict__ Ew) {
  const int idx = blockIdx.x * blockDim.x + threadIdx.x;
  if (idx >= 2049 * 64) return;
  const int row = idx >> 6, d = idx & 63;
  float acc = 0.0f;
#pragma unroll 4
  for (int e = 0; e < 64; ++e) acc += rel_embed[row * 64 + e] * rel_w[e * 64 + d];
  Ew[idx] = (__bf16)acc;
}

// ---------------------------------------------------------------------------
// Fused flash attention with relative-position bias.
// Block = 128 thr (4 waves); block handles (b, h, 64 q-rows); wave owns 16 q.
// scores = (Q.K^T + Q.Ew[k-q+L]) / 8, masked, online softmax, O = P.V.
// K tile [k][d] and pre-transposed V tile [d][k] staged with async-to-LDS,
// double-buffered across the k loop.
// ---------------------------------------------------------------------------
__global__ __launch_bounds__(128) void attn_rel_kernel(
    const __bf16* __restrict__ Qg, const __bf16* __restrict__ Kg,
    const __bf16* __restrict__ Vt_g, const __bf16* __restrict__ Ew,
    const unsigned char* __restrict__ mask, __bf16* __restrict__ Og) {
  __shared__ __align__(16) __bf16 Ks[2][32][72];  // k x d
  __shared__ __align__(16) __bf16 Vt[2][64][40];  // d x k
  __shared__ float  Rs[4][16][48];
  __shared__ __align__(16) __bf16 Ps[4][16][32];

  const int tid  = threadIdx.x;
  const int lane = tid & 31;
  const int w    = tid >> 5;
  const int r    = lane & 15;
  const int hf   = lane >> 4;

  const int qb = blockIdx.x & 15;
  const int h  = (blockIdx.x >> 4) & 15;
  const int b  = blockIdx.x >> 8;
  const int q0 = qb * 64 + w * 16;

  const __bf16* gk = Kg + ((size_t)b * L_SEQ) * FDIM + h * DHEAD;   // [k][d]
  const __bf16* gv = Vt_g + (((size_t)b * 16 + h) * 64) * L_SEQ;    // [d][k]

  auto issue_tiles = [&](int buf, int k0) {
    // K tile: 32 rows x 8 16B-chunks = 256, 2 per thread
#pragma unroll
    for (int it = 0; it < 2; ++it) {
      const int idx = it * 128 + tid;
      const int rr = idx >> 3, cc = idx & 7;
      async_ld_b128(&Ks[buf][rr][cc * 8],
                    gk + (size_t)(k0 + rr) * FDIM + cc * 8);
    }
    // V tile: 64 d-rows x 4 chunks = 256, 2 per thread
#pragma unroll
    for (int it = 0; it < 2; ++it) {
      const int idx = it * 128 + tid;
      const int rr = idx >> 2, cc = idx & 3;
      async_ld_b128(&Vt[buf][rr][cc * 8],
                    gv + (size_t)rr * L_SEQ + k0 + cc * 8);
    }
  };

  // Q fragments for this wave's 16 rows (d contraction split into 2x32)
  v16bf qa[2];
  const __bf16* qbase = Qg + ((size_t)(b * L_SEQ + q0)) * FDIM + h * DHEAD;
#pragma unroll
  for (int c = 0; c < 2; ++c)
    qa[c] = ld_afrag(qbase + (size_t)r * FDIM + c * 32, hf);

  float m_i[8], l_i[8];
  v8f o[4] = {};
#pragma unroll
  for (int i = 0; i < 8; ++i) { m_i[i] = -3.0e38f; l_i[i] = 0.0f; }

  int buf = 0;
  issue_tiles(0, 0);
#pragma unroll 1
  for (int k0 = 0; k0 < L_SEQ; k0 += 32) {
    wait_async0();
    __syncthreads();
    if (k0 + 32 < L_SEQ) issue_tiles(buf ^ 1, k0 + 32);

    // S = Q @ K^T  (16 x 32)
    v8f s[2] = {};
#pragma unroll
    for (int f = 0; f < 2; ++f)
#pragma unroll
      for (int c = 0; c < 2; ++c) {
        const v16bf kb = ld_bfrag(&Ks[buf][f * 16 + r][c * 32], hf);
        s[f] = wmma_bf16(qa[c], kb, s[f]);
      }

    // R = Q @ EwWin^T  (16 x 48): rel bias over the diagonal band
    const int base = k0 - q0 - 15 + L_SEQ;       // always in [1, 2001]
    v8f rf[3] = {};
#pragma unroll
    for (int cs = 0; cs < 3; ++cs)
#pragma unroll
      for (int c = 0; c < 2; ++c) {
        const v16bf eb =
            ld_bfrag(Ew + (size_t)(base + cs * 16 + r) * DHEAD + c * 32, hf);
        rf[cs] = wmma_bf16(qa[c], eb, rf[cs]);
      }
#pragma unroll
    for (int cs = 0; cs < 3; ++cs)
#pragma unroll
      for (int i = 0; i < 8; ++i) Rs[w][i + 8 * hf][cs * 16 + r] = rf[cs][i];
    // same-wave LDS ops are in-order: shift-gather needs no barrier

    // combine + scale(1/sqrt(64)) + mask
#pragma unroll
    for (int f = 0; f < 2; ++f)
#pragma unroll
      for (int i = 0; i < 8; ++i) {
        const int ql = i + 8 * hf;
        const int kl = r + 16 * f;
        float v = (s[f][i] + Rs[w][ql][kl - ql + 15]) * 0.125f;
        const size_t midx = ((size_t)(b * L_SEQ + q0 + ql)) * L_SEQ + (k0 + kl);
        if (!mask[midx]) v = -3.4028235e38f;
        s[f][i] = v;
      }

    // online softmax update (row reductions across 16 lanes of each half)
#pragma unroll
    for (int i = 0; i < 8; ++i) {
      float rm = hmax16(fmaxf(s[0][i], s[1][i]));
      const float mn = fmaxf(m_i[i], rm);
      const float p0 = __expf(s[0][i] - mn);
      const float p1 = __expf(s[1][i] - mn);
      s[0][i] = p0; s[1][i] = p1;
      const float rs = hsum16(p0 + p1);
      const float sc = __expf(m_i[i] - mn);
      l_i[i] = l_i[i] * sc + rs;
      m_i[i] = mn;
#pragma unroll
      for (int j = 0; j < 4; ++j) o[j][i] *= sc;
    }

    // P tile -> LDS (bf16) -> A fragment (C-layout -> A-layout reshape)
#pragma unroll
    for (int f = 0; f < 2; ++f)
#pragma unroll
      for (int i = 0; i < 8; ++i)
        Ps[w][i + 8 * hf][f * 16 + r] = (__bf16)s[f][i];
    const v16bf pa = ld_afrag(&Ps[w][r][0], hf);

    // O += P @ V  (16 x 64), V-frags contiguous from transposed tile
#pragma unroll
    for (int j = 0; j < 4; ++j) {
      const v16bf vb = ld_bfrag(&Vt[buf][j * 16 + r][0], hf);
      o[j] = wmma_bf16(pa, vb, o[j]);
    }
    buf ^= 1;
  }

  // normalize + store bf16 attention output
#pragma unroll
  for (int i = 0; i < 8; ++i) {
    const float inv = 1.0f / l_i[i];
    const int row = q0 + i + 8 * hf;
#pragma unroll
    for (int j = 0; j < 4; ++j)
      Og[((size_t)(b * L_SEQ + row)) * FDIM + h * DHEAD + j * 16 + r] =
          (__bf16)(o[j][i] * inv);
  }
}

// ---------------------------------------------------------------------------
extern "C" void kernel_launch(void* const* d_in, const int* in_sizes, int n_in,
                              void* d_out, int out_size, void* d_ws, size_t ws_size,
                              hipStream_t stream) {
  (void)in_sizes; (void)n_in; (void)out_size; (void)ws_size;
  const float* kv  = (const float*)d_in[0];
  const float* q   = (const float*)d_in[1];
  const unsigned char* mask = (const unsigned char*)d_in[2];
  const float* Wq  = (const float*)d_in[3];
  const float* bq  = (const float*)d_in[4];
  const float* Wk  = (const float*)d_in[5];
  const float* bk  = (const float*)d_in[6];
  const float* Wv  = (const float*)d_in[7];
  const float* bv  = (const float*)d_in[8];
  const float* Wo  = (const float*)d_in[9];
  const float* bo  = (const float*)d_in[10];
  const float* rel_embed = (const float*)d_in[11];
  const float* rel_w     = (const float*)d_in[12];

  char* ws = (char*)d_ws;
  const size_t MB = 1024 * 1024;
  __bf16* Qb  = (__bf16*)(ws);             // 8 MB  Q bf16 [B*L, F]
  __bf16* Kb  = (__bf16*)(ws + 8 * MB);    // 8 MB  K bf16 [B*L, F]
  __bf16* Vb2 = (__bf16*)(ws + 16 * MB);   // 8 MB  V bf16 [B,H,D,L]
  __bf16* Ab  = (__bf16*)(ws + 24 * MB);   // 8 MB  attn out bf16 [B*L, F]
  __bf16* qX  = (__bf16*)(ws + 32 * MB);   // 8 MB  q  bf16
  __bf16* kvX = (__bf16*)(ws + 40 * MB);   // 8 MB  kv bf16
  __bf16* Wqt = (__bf16*)(ws + 48 * MB);   // 2 MB  Wq^T bf16
  __bf16* Wkt = (__bf16*)(ws + 50 * MB);   // 2 MB
  __bf16* Wvt = (__bf16*)(ws + 52 * MB);   // 2 MB
  __bf16* Wot = (__bf16*)(ws + 54 * MB);   // 2 MB
  __bf16* Ewb = (__bf16*)(ws + 56 * MB);   // 2049*64*2 B

  const int M = 4096, N = 1024, K = 1024;
  const int NTOT = M * N;                        // 4M elements
  const int gemm_blocks = (M / 128) * (N / 64);  // 512
  const dim3 tgrid(32, 32);

  cvt_bf16_kernel<<<NTOT / 4 / 256, 256, 0, stream>>>(q,  qX,  NTOT);
  cvt_bf16_kernel<<<NTOT / 4 / 256, 256, 0, stream>>>(kv, kvX, NTOT);
  transpose_bf16_kernel<<<tgrid, 256, 0, stream>>>(Wq, Wqt);
  transpose_bf16_kernel<<<tgrid, 256, 0, stream>>>(Wk, Wkt);
  transpose_bf16_kernel<<<tgrid, 256, 0, stream>>>(Wv, Wvt);
  transpose_bf16_kernel<<<tgrid, 256, 0, stream>>>(Wo, Wot);
  ew_kernel<<<(2049 * 64 + 255) / 256, 256, 0, stream>>>(rel_embed, rel_w, Ewb);

  gemm_bias_kernel<0><<<gemm_blocks, 128, 0, stream>>>(qX,  Wqt, bq, Qb,  M, N, K);
  gemm_bias_kernel<0><<<gemm_blocks, 128, 0, stream>>>(kvX, Wkt, bk, Kb,  M, N, K);
  gemm_bias_kernel<1><<<gemm_blocks, 128, 0, stream>>>(kvX, Wvt, bv, Vb2, M, N, K);
  attn_rel_kernel<<<1024, 128, 0, stream>>>(Qb, Kb, Vb2, Ewb, mask, Ab);
  gemm_bias_kernel<2><<<gemm_blocks, 128, 0, stream>>>(Ab, Wot, bo, d_out, M, N, K);
}
